// SingleHeadAttention_49074296324547
// MI455X (gfx1250) — matrix operand order repaired and checked by reading, not compiled, not run on hardware
//
#include <hip/hip_runtime.h>
#include <math.h>

typedef __bf16 bf16_t;
typedef bf16_t v16bf __attribute__((ext_vector_type(16)));
typedef bf16_t v8bf  __attribute__((ext_vector_type(8)));
typedef float  v8f   __attribute__((ext_vector_type(8)));
typedef float  v4f   __attribute__((ext_vector_type(4)));

#define WMMA_BF16(A_, B_, C_) \
  __builtin_amdgcn_wmma_f32_16x16x32_bf16(false, (A_), false, (B_), (short)0, (C_), false, false)

constexpr int Bn = 4, Tn = 4096, Cn = 512, Hn = 64;
constexpr float SCALE = 0.04419417382415922f; // 512^-0.5

// float -> bf16 via native conversion (RNE on gfx1250 hardware cvt)
static __device__ __forceinline__ bf16_t f2bf(float f) { return (bf16_t)f; }

static __device__ __forceinline__ v16bf combine8(v8bf lo, v8bf hi) {
  return __builtin_shufflevector(lo, hi, 0,1,2,3,4,5,6,7,8,9,10,11,12,13,14,15);
}

// ---------------- W transpose + bf16 convert: [C,H] f32 -> [H,C] bf16 ----------------
__global__ void wt_kernel(const float* __restrict__ Wk, const float* __restrict__ Wq,
                          const float* __restrict__ Wv,
                          bf16_t* __restrict__ Wtk, bf16_t* __restrict__ Wtq,
                          bf16_t* __restrict__ Wtv) {
  int idx = blockIdx.x * blockDim.x + threadIdx.x;
  if (idx >= Cn * Hn) return;
  int c = idx >> 6, h = idx & 63;
  Wtk[h * Cn + c] = f2bf(Wk[idx]);
  Wtq[h * Cn + c] = f2bf(Wq[idx]);
  Wtv[h * Cn + c] = f2bf(Wv[idx]);
}

// ---------------- Projections: Q,K [B*T,H] bf16 row-major; V transposed [B,H,T] bf16 ----------------
__global__ __launch_bounds__(128) void proj_kernel(
    const float* __restrict__ x,
    const bf16_t* __restrict__ Wtq, const bf16_t* __restrict__ Wtk, const bf16_t* __restrict__ Wtv,
    bf16_t* __restrict__ Q, bf16_t* __restrict__ K, bf16_t* __restrict__ Vt) {
  const int lane = threadIdx.x & 31;
  const int wave = threadIdx.x >> 5;
  const int tile = blockIdx.x * 4 + wave;      // 0..1023, 16 rows each over B*T
  const int t0   = tile * 16;
  const int col  = lane & 15;                  // also A-row index
  const bool hi  = (lane >= 16);
  const int kbA  = hi ? 8 : 0;                 // unused; A loads from kbB like B
  const int kbB  = hi ? 16 : 0;                // B-fragment K base
  const int rb   = hi ? 8 : 0;                 // C/D row base

  v8f accQ[4] = {}, accK[4] = {}, accV[4] = {};
  const float* xrow = x + (size_t)(t0 + col) * Cn;

  for (int k0 = 0; k0 < Cn; k0 += 32) {
    v4f f0 = *(const v4f*)(xrow + k0 + kbB);
    v4f f1 = *(const v4f*)(xrow + k0 + kbB + 4);
    v4f f2 = *(const v4f*)(xrow + k0 + kbB + 8);
    v4f f3 = *(const v4f*)(xrow + k0 + kbB + 12);
    v16bf a;
#pragma unroll
    for (int i = 0; i < 4; ++i) {
      a[i]      = f2bf(f0[i]);
      a[4 + i]  = f2bf(f1[i]);
      a[8 + i]  = f2bf(f2[i]);
      a[12 + i] = f2bf(f3[i]);
    }
#pragma unroll
    for (int f = 0; f < 4; ++f) {
      const int n = f * 16 + col;
      v16bf bq = *(const v16bf*)(Wtq + (size_t)n * Cn + k0 + kbB);
      v16bf bk = *(const v16bf*)(Wtk + (size_t)n * Cn + k0 + kbB);
      v16bf bv = *(const v16bf*)(Wtv + (size_t)n * Cn + k0 + kbB);
      accQ[f] = WMMA_BF16(a, bq, accQ[f]);
      accK[f] = WMMA_BF16(a, bk, accK[f]);
      accV[f] = WMMA_BF16(a, bv, accV[f]);
    }
  }

#pragma unroll
  for (int f = 0; f < 4; ++f) {
    const int h = f * 16 + col;
#pragma unroll
    for (int r = 0; r < 8; ++r) {
      const int rr = t0 + r + rb;              // global row in [0, B*T)
      Q[(size_t)rr * Hn + h] = f2bf(accQ[f][r]);
      K[(size_t)rr * Hn + h] = f2bf(accK[f][r]);
      const int bb = rr >> 12;                 // T = 4096
      const int tt = rr & (Tn - 1);
      Vt[((size_t)(bb * Hn + h)) * Tn + tt] = f2bf(accV[f][r]);
    }
  }
}

// ---------------- Flash attention, split-K: 4 waves cooperate on one 16-query tile ----------------
__global__ __launch_bounds__(128) void attn_kernel(
    const bf16_t* __restrict__ Q, const bf16_t* __restrict__ K,
    const bf16_t* __restrict__ Vt, float* __restrict__ out) {
  __shared__ __align__(64) bf16_t pt[4][16 * 32];  // per-wave P tile (16x32 bf16)
  __shared__ float sm[4][16];                      // per-wave row-max partials
  __shared__ float sl[4][16];                      // per-wave row-sum partials
  __shared__ float sacc[4][16][64];                // per-wave O partials (16x64 f32)

  const int tid  = threadIdx.x;
  const int lane = tid & 31;
  const int wid  = tid >> 5;                   // 0..3: key-split index
  const int tile = blockIdx.x;                 // 0..1023 over B*(T/16)
  const int b    = tile >> 8;                  // T/16 = 256 tiles per batch
  const int q0   = (tile & 255) * 16;          // query row base within batch
  const int row  = lane & 15;
  const int col  = lane & 15;
  const bool hi  = (lane >= 16);
  const int kbA  = hi ? 8 : 0;
  const int kbB  = hi ? 16 : 0;
  const int rb   = hi ? 8 : 0;

  const bf16_t* Qb = Q + (size_t)b * Tn * Hn;
  const bf16_t* Kb = K + (size_t)b * Tn * Hn;
  const bf16_t* Vb = Vt + (size_t)b * Hn * Tn;

  // Q A-fragments covering head dim K = 0..31 and 32..63 (same in all 4 waves)
  v16bf qa0, qa1;
  {
    const bf16_t* qp = Qb + (size_t)(q0 + row) * Hn;
    qa0 = combine8(*(const v8bf*)(qp + kbB),      *(const v8bf*)(qp + 8 + kbB));
    qa1 = combine8(*(const v8bf*)(qp + 32 + kbB), *(const v8bf*)(qp + 40 + kbB));
  }

  v8f acc[4] = {};                             // O partial accumulator over H
  float m[8], l[8];
#pragma unroll
  for (int r = 0; r < 8; ++r) { m[r] = -__builtin_inff(); l[r] = 0.0f; }

  const int nblk = (q0 + 15) / 32 + 1;         // causal: key blocks with s0 <= q0+15
  for (int blk = wid; blk < nblk; blk += 4) {  // split-K: each wave takes every 4th block
    const int s0 = blk * 32;

    // prefetch this wave's next key/value block (128 rows ahead)
    if (blk + 4 < nblk) {
      __builtin_prefetch(Kb + (size_t)(s0 + 128 + col) * Hn, 0, 3);
      __builtin_prefetch(Vb + (size_t)col * Tn + s0 + 128, 0, 3);
    }

    // ---- scores S[16x32] = Q Kt, two 16x16 f32 fragments ----
    v8f sc0 = {}, sc1 = {};
    {
      const bf16_t* kp0 = Kb + (size_t)(s0 + col) * Hn;
      const bf16_t* kp1 = Kb + (size_t)(s0 + 16 + col) * Hn;
      v16bf b00 = *(const v16bf*)(kp0 + kbB);
      v16bf b01 = *(const v16bf*)(kp0 + 32 + kbB);
      v16bf b10 = *(const v16bf*)(kp1 + kbB);
      v16bf b11 = *(const v16bf*)(kp1 + 32 + kbB);
      sc0 = WMMA_BF16(qa0, b00, sc0);
      sc0 = WMMA_BF16(qa1, b01, sc0);
      sc1 = WMMA_BF16(qa0, b10, sc1);
      sc1 = WMMA_BF16(qa1, b11, sc1);
    }

    // ---- scale + causal mask + online softmax ----
    float mloc[8];
#pragma unroll
    for (int r = 0; r < 8; ++r) {
      const int trow = q0 + r + rb;
      float a0 = sc0[r] * SCALE;
      float a1 = sc1[r] * SCALE;
      if (s0 + col > trow)      a0 = -__builtin_inff();
      if (s0 + 16 + col > trow) a1 = -__builtin_inff();
      sc0[r] = a0; sc1[r] = a1;
      mloc[r] = fmaxf(a0, a1);
    }
#pragma unroll
    for (int xm = 1; xm < 16; xm <<= 1) {
#pragma unroll
      for (int r = 0; r < 8; ++r)
        mloc[r] = fmaxf(mloc[r], __shfl_xor(mloc[r], xm, 32));
    }

    float corr[8], rsum[8];
#pragma unroll
    for (int r = 0; r < 8; ++r) {
      const float mnew = fmaxf(m[r], mloc[r]);
      corr[r] = __expf(m[r] - mnew);
      m[r] = mnew;
      const float p0 = __expf(sc0[r] - mnew);
      const float p1 = __expf(sc1[r] - mnew);
      sc0[r] = p0; sc1[r] = p1;
      rsum[r] = p0 + p1;
    }
#pragma unroll
    for (int xm = 1; xm < 16; xm <<= 1) {
#pragma unroll
      for (int r = 0; r < 8; ++r)
        rsum[r] += __shfl_xor(rsum[r], xm, 32);
    }
#pragma unroll
    for (int r = 0; r < 8; ++r) l[r] = l[r] * corr[r] + rsum[r];
#pragma unroll
    for (int f = 0; f < 4; ++f)
#pragma unroll
      for (int r = 0; r < 8; ++r) acc[f][r] *= corr[r];

    // ---- relayout P (C-layout) -> A-fragment via per-wave LDS tile ----
    bf16_t* pl = pt[wid];
#pragma unroll
    for (int r = 0; r < 8; ++r) {
      const int rr = r + rb;
      pl[rr * 32 + col]      = f2bf(sc0[r]);
      pl[rr * 32 + 16 + col] = f2bf(sc1[r]);
    }
    asm volatile("s_wait_dscnt 0" ::: "memory");   // per-wave DS ordering guarantee
    v16bf pa = combine8(*(const v8bf*)(pl + row * 32 + kbB),
                        *(const v8bf*)(pl + row * 32 + 8 + kbB));

    // ---- O += P @ V   (B fragments contiguous thanks to V^T layout) ----
#pragma unroll
    for (int f = 0; f < 4; ++f) {
      const bf16_t* vp = Vb + (size_t)(f * 16 + col) * Tn + s0;
      v16bf bfrag = *(const v16bf*)(vp + kbB);
      acc[f] = WMMA_BF16(pa, bfrag, acc[f]);
    }
  }

  // ---- publish per-wave partials ----
  if (col == 0) {                              // lanes 0 (rows 0-7) and 16 (rows 8-15)
#pragma unroll
    for (int r = 0; r < 8; ++r) { sm[wid][r + rb] = m[r]; sl[wid][r + rb] = l[r]; }
  }
#pragma unroll
  for (int f = 0; f < 4; ++f)
#pragma unroll
    for (int r = 0; r < 8; ++r)
      sacc[wid][r + rb][f * 16 + col] = acc[f][r];
  __syncthreads();

  // ---- split-K combine + normalize + store (128 threads over 16x64 outputs) ----
#pragma unroll
  for (int j = 0; j < 8; ++j) {
    const int e  = j * 128 + tid;              // 0..1023
    const int rr = e >> 6;                     // 0..15
    const int h  = e & 63;
    const float M = fmaxf(fmaxf(sm[0][rr], sm[1][rr]), fmaxf(sm[2][rr], sm[3][rr]));
    float L = 0.0f, val = 0.0f;
#pragma unroll
    for (int w = 0; w < 4; ++w) {
      const float s = __expf(sm[w][rr] - M);   // 0 for waves that processed no blocks
      L   += sl[w][rr] * s;
      val += sacc[w][rr][h] * s;
    }
    out[((size_t)(b * Tn + q0 + rr)) * Hn + h] = val / L;
  }
}

extern "C" void kernel_launch(void* const* d_in, const int* in_sizes, int n_in,
                              void* d_out, int out_size, void* d_ws, size_t ws_size,
                              hipStream_t stream) {
  (void)in_sizes; (void)n_in; (void)out_size; (void)ws_size;
  const float* x  = (const float*)d_in[0];
  const float* Wk = (const float*)d_in[1];
  const float* Wq = (const float*)d_in[2];
  const float* Wv = (const float*)d_in[3];
  float* out = (float*)d_out;

  char* ws = (char*)d_ws;
  bf16_t* Wtq = (bf16_t*)(ws + 0);
  bf16_t* Wtk = (bf16_t*)(ws + (64 << 10));
  bf16_t* Wtv = (bf16_t*)(ws + (128 << 10));
  bf16_t* Qb  = (bf16_t*)(ws + (256 << 10));                       // 2 MB
  bf16_t* Kb  = (bf16_t*)(ws + (256 << 10) + (2u << 20));          // 2 MB
  bf16_t* Vt  = (bf16_t*)(ws + (256 << 10) + (4u << 20));          // 2 MB [B,H,T]

  wt_kernel<<<(Cn * Hn + 255) / 256, 256, 0, stream>>>(Wk, Wq, Wv, Wtk, Wtq, Wtv);
  proj_kernel<<<(Bn * Tn / 16) / 4, 128, 0, stream>>>(x, Wtq, Wtk, Wtv, Qb, Kb, Vt);
  attn_kernel<<<Bn * (Tn / 16), 128, 0, stream>>>(Qb, Kb, Vt, out);
}
